// Baseline_SA_17815524343884
// MI455X (gfx1250) — matrix operand order repaired and checked by reading, not compiled, not run on hardware
//
#include <hip/hip_runtime.h>
#include <hip/hip_bf16.h>
#include <stdint.h>

typedef __attribute__((ext_vector_type(16))) _Float16 v16h;
typedef __attribute__((ext_vector_type(8)))  _Float16 v8h;
typedef __attribute__((ext_vector_type(8)))  float    v8f;

#define B_  64
#define T_  100
#define N_  20
#define C_  1024
#define F_  256
#define H_  256
#define L_  512
#define NWG 32

// ---------- helpers ----------

__device__ __forceinline__ unsigned lds_u32(const void* p) {
  // generic -> LDS(addrspace 3) -> 32-bit LDS byte offset
  return (unsigned)(uintptr_t)(__attribute__((address_space(3))) const char*)p;
}

__device__ __forceinline__ v8f wmma_f16(v16h a, v16h b, v8f c) {
  return __builtin_amdgcn_wmma_f32_16x16x32_f16(false, a, false, b, (short)0, c,
                                                false, false);
}

__device__ __forceinline__ float sigmoidf_(float x) {
  return 1.0f / (1.0f + __expf(-x));
}

// One 16x16 output tile, K-loop over global f32 A and global f16 B^T.
// A: row-major f32, rows rowbase..rowbase+15, leading dim lda.
// Bt: row n holds the contraction axis contiguously, leading dim ldb.
__device__ __forceinline__ v8f tile_mm_g(const float* __restrict__ A, int rowbase,
                                         int lda, const _Float16* __restrict__ Bt,
                                         int colbase, int ldb, int K) {
  const int lane = threadIdx.x & 31;
  const int m = lane & 15, grp = lane >> 4;
  const float*    ar = A  + (size_t)(rowbase + m) * lda + grp * 8;
  const _Float16* br = Bt + (size_t)(colbase + m) * ldb + grp * 8;
  v8f acc = {0.f, 0.f, 0.f, 0.f, 0.f, 0.f, 0.f, 0.f};
  for (int kc = 0; kc < K; kc += 32) {
    float4 f0 = *(const float4*)(ar + kc);
    float4 f1 = *(const float4*)(ar + kc + 4);
    float4 f2 = *(const float4*)(ar + kc + 16);
    float4 f3 = *(const float4*)(ar + kc + 20);
    v16h a;
    a[0]=(_Float16)f0.x;  a[1]=(_Float16)f0.y;  a[2]=(_Float16)f0.z;  a[3]=(_Float16)f0.w;
    a[4]=(_Float16)f1.x;  a[5]=(_Float16)f1.y;  a[6]=(_Float16)f1.z;  a[7]=(_Float16)f1.w;
    a[8]=(_Float16)f2.x;  a[9]=(_Float16)f2.y;  a[10]=(_Float16)f2.z; a[11]=(_Float16)f2.w;
    a[12]=(_Float16)f3.x; a[13]=(_Float16)f3.y; a[14]=(_Float16)f3.z; a[15]=(_Float16)f3.w;
    v8h b0 = *(const v8h*)(br + kc);
    v8h b1 = *(const v8h*)(br + kc + 16);
    v16h b = __builtin_shufflevector(b0, b1, 0,1,2,3,4,5,6,7,8,9,10,11,12,13,14,15);
    acc = wmma_f16(a, b, acc);
  }
  return acc;
}

// ---------- big feed-forward GEMMs ----------
// Workgroup = 8 waves, each wave owns one 16-row strip and all 16 col tiles
// (H = 256). Weight chunk (256 x 32 f16) is staged into LDS with CDNA5
// async global->LDS copies each K step; A (f32) is read straight from HBM and
// converted to f16 in registers.  MODE 0: obj (BN folded, relu, mask fused);
// MODE 1: image_part; MODE 2: img projection.
template <int MODE>
__global__ void __launch_bounds__(256)
wmma_gemm(const float* __restrict__ A, const _Float16* __restrict__ Bt,
          const float* __restrict__ bias, float* __restrict__ Out,
          float* __restrict__ maskbuf, int K, int lda) {
  __shared__ _Float16 ldsB[256][40];  // 32 halfs payload + pad (80B row, 16B aligned)
  const int tid = threadIdx.x;
  const int wave = tid >> 5, lane = tid & 31;
  const int m = lane & 15, grp = lane >> 4;
  const int rowbase = (blockIdx.x * 8 + wave) * 16;

  const float* ar = A + (size_t)(rowbase + m) * lda + grp * 8;
  const _Float16* bsrc = Bt + (size_t)tid * K;  // thread stages weight row 'tid'
  const unsigned ldst = lds_u32(&ldsB[tid][0]);

  v8f acc[16];
#pragma unroll
  for (int i = 0; i < 16; ++i) acc[i] = (v8f){0.f,0.f,0.f,0.f,0.f,0.f,0.f,0.f};
  float rowsum = 0.f;

  for (int kc = 0; kc < K; kc += 32) {
    __syncthreads();  // previous chunk fully consumed
    // async-stage 256x32 f16 weight chunk into LDS (ASYNCcnt path)
#pragma unroll
    for (int q = 0; q < 4; ++q) {
      unsigned long long ga = (unsigned long long)(uintptr_t)(bsrc + kc + q * 8);
      unsigned la = ldst + q * 16;
      asm volatile("global_load_async_to_lds_b128 %0, %1, off"
                   :: "v"(la), "v"(ga) : "memory");
    }
    // A fragment (overlaps with async copy)
    float4 f0 = *(const float4*)(ar + kc);
    float4 f1 = *(const float4*)(ar + kc + 4);
    float4 f2 = *(const float4*)(ar + kc + 16);
    float4 f3 = *(const float4*)(ar + kc + 20);
    v16h a;
    a[0]=(_Float16)f0.x;  a[1]=(_Float16)f0.y;  a[2]=(_Float16)f0.z;  a[3]=(_Float16)f0.w;
    a[4]=(_Float16)f1.x;  a[5]=(_Float16)f1.y;  a[6]=(_Float16)f1.z;  a[7]=(_Float16)f1.w;
    a[8]=(_Float16)f2.x;  a[9]=(_Float16)f2.y;  a[10]=(_Float16)f2.z; a[11]=(_Float16)f2.w;
    a[12]=(_Float16)f3.x; a[13]=(_Float16)f3.y; a[14]=(_Float16)f3.z; a[15]=(_Float16)f3.w;
    if (MODE == 0)
      rowsum += (f0.x+f0.y+f0.z+f0.w) + (f1.x+f1.y+f1.z+f1.w) +
                (f2.x+f2.y+f2.z+f2.w) + (f3.x+f3.y+f3.z+f3.w);
    asm volatile("s_wait_asynccnt 0" ::: "memory");
    __syncthreads();  // all waves' async copies landed
#pragma unroll
    for (int ct = 0; ct < 16; ++ct) {
      const _Float16* bp = &ldsB[ct * 16 + m][grp * 8];
      v8h b0 = *(const v8h*)bp;
      v8h b1 = *(const v8h*)(bp + 16);
      v16h b = __builtin_shufflevector(b0, b1, 0,1,2,3,4,5,6,7,8,9,10,11,12,13,14,15);
      acc[ct] = wmma_f16(a, b, acc[ct]);
    }
  }

  // mask (mode 0): lane l holds partial for row (l&15); combine halves
  float mk = 0.f;
  if (MODE == 0) {
    float rs = rowsum + __shfl_xor(rowsum, 16, 32);
    mk = (rs != 0.f) ? 1.f : 0.f;
    if (lane < 16) {
      int gr = rowbase + lane;
      int b = gr / (T_ * N_); int rem = gr % (T_ * N_);
      int tt = rem / N_;      int n  = rem % N_;
      maskbuf[(tt * N_ + n) * B_ + b] = mk;
    }
  }

#pragma unroll
  for (int ct = 0; ct < 16; ++ct) {
    int col = ct * 16 + m;
    float bcol = bias[col];
#pragma unroll
    for (int r = 0; r < 8; ++r) {
      int row = r + grp * 8;
      int gr = rowbase + row;
      float v = acc[ct][r] + bcol;
      if (MODE == 0) {
        float mrow = __shfl(mk, row, 32);
        v = fmaxf(v, 0.f) * mrow;
        int b = gr / (T_ * N_); int rem = gr % (T_ * N_);
        int tt = rem / N_;      int n  = rem % N_;
        Out[((size_t)(tt * N_ + n) * B_ + b) * H_ + col] = v;
      } else if (MODE == 1) {
        Out[(size_t)gr * H_ + col] = v;
      } else {
        int b = gr / T_, tt = gr % T_;
        Out[((size_t)tt * B_ + b) * H_ + col] = v;
      }
    }
  }
}

// ---------- persistent recurrent kernel ----------

__device__ __forceinline__ void grid_sync(int* bar) {
  __syncthreads();
  if (threadIdx.x == 0) {
    __threadfence();
    volatile int* gen = bar + 1;
    int g = *gen;
    if (atomicAdd(bar, 1) == NWG - 1) {
      bar[0] = 0;
      __threadfence();
      atomicAdd((int*)gen, 1);
    } else {
      while (*gen == g) __builtin_amdgcn_s_sleep(1);
    }
  }
  __syncthreads();
  __threadfence();
}

__global__ void __launch_bounds__(256)
recurrent_k(const float* __restrict__ ip, const float* __restrict__ obj,
            const float* __restrict__ img, const float* __restrict__ maskbuf,
            const _Float16* __restrict__ Wih, const _Float16* __restrict__ Whh,
            const _Float16* __restrict__ Wwa, const float* __restrict__ gsum,
            const float* __restrict__ attw, const float* __restrict__ outW,
            const float* __restrict__ outB, float* hx, float* cx, float* hterm,
            float* pa, float* fusion, float* out_alphas, float* out_stack,
            int* bar) {
  __shared__ float shg[4 * 16 * 64];
  __shared__ float sh_sc[20], sh_al[20];
  const int tid = threadIdx.x, wave = tid >> 5, lane = tid & 31;
  const int m = lane & 15, grp = lane >> 4;
  const int wg = blockIdx.x;

  for (int t = 0; t < T_; ++t) {
    // ---- Phase A: hterm = hx@Whh^T (64x2048), pa = hx@att_wa (64x256) ----
    for (int j = wg * 8 + wave; j < 576; j += NWG * 8) {
      if (j < 512) {
        int mt = j & 3, ct = j >> 2;
        v8f acc = tile_mm_g(hx, mt * 16, L_, Whh, ct * 16, L_, L_);
#pragma unroll
        for (int r = 0; r < 8; ++r)
          hterm[(size_t)(mt * 16 + r + grp * 8) * 2048 + ct * 16 + m] = acc[r];
      } else {
        int j2 = j - 512, mt = j2 & 3, ct = j2 >> 2;
        v8f acc = tile_mm_g(hx, mt * 16, L_, Wwa, ct * 16, L_, L_);
#pragma unroll
        for (int r = 0; r < 8; ++r)
          pa[(mt * 16 + r + grp * 8) * H_ + ct * 16 + m] = acc[r];
      }
    }
    if (wg == 0 && t > 0 && tid < 128) {  // pred for step t-1 from current hx
      int b = tid >> 1, c = tid & 1;
      float s = outB[c];
      for (int l = 0; l < L_; ++l) s += hx[b * L_ + l] * outW[l * 2 + c];
      out_stack[(t - 1) * (B_ * 2) + b * 2 + c] = s;
    }
    grid_sync(bar);

    // ---- Phase B: attention (workgroup-local softmax per batch element) ----
    for (int bb = 0; bb < 2; ++bb) {
      int b = wg + bb * NWG;
      for (int n = wave; n < N_; n += 8) {
        float p = 0.f;
        const float* iprow = ip + ((size_t)(t * N_ + n) * B_ + b) * H_;
        for (int h = lane; h < H_; h += 32)
          p += fmaxf(pa[b * H_ + h] + iprow[h], 0.f) * attw[h];
#pragma unroll
        for (int off = 16; off; off >>= 1) p += __shfl_xor(p, off, 32);
        if (lane == 0) sh_sc[n] = p;
      }
      __syncthreads();
      if (tid == 0) {
        float mx = -1e30f;
        for (int n = 0; n < N_; ++n) mx = fmaxf(mx, sh_sc[n]);
        float s = 0.f;
        for (int n = 0; n < N_; ++n) { float e = __expf(sh_sc[n] - mx); sh_al[n] = e; s += e; }
        float inv = 1.f / s;
        for (int n = 0; n < N_; ++n) {
          float al = sh_al[n] * inv * maskbuf[(t * N_ + n) * B_ + b];
          sh_al[n] = al;
          out_alphas[(size_t)b * (T_ * N_) + t * N_ + n] = al;
        }
      }
      __syncthreads();
      {
        int h = tid;  // H_ == 256 == blockDim
        float att = 0.f;
        for (int n = 0; n < N_; ++n)
          att += sh_al[n] * obj[((size_t)(t * N_ + n) * B_ + b) * H_ + h];
        fusion[b * (2 * H_) + H_ + h] = att;
        fusion[b * (2 * H_) + h]      = img[((size_t)t * B_ + b) * H_ + h];
      }
      __syncthreads();
    }
    grid_sync(bar);

    // ---- Phase C: gates = fusion@Wih^T + hterm + (b_ih+b_hh); LSTM update ----
    {
      int btile = wg & 3, jb = wg >> 2;
      for (int f = wave; f < 16; f += 8) {
        int q = f >> 2, jt = f & 3;
        int colbase = q * L_ + jb * 64 + jt * 16;
        v8f acc = tile_mm_g(fusion, btile * 16, 2 * H_, Wih, colbase, 2 * H_, 2 * H_);
#pragma unroll
        for (int r = 0; r < 8; ++r) {
          int mm = r + grp * 8;
          int col = colbase + m;
          int row = btile * 16 + mm;
          shg[(q * 16 + mm) * 64 + jt * 16 + m] =
              acc[r] + gsum[col] + hterm[(size_t)row * 2048 + col];
        }
      }
      __syncthreads();
      for (int e = tid; e < 1024; e += 256) {
        int mm = e >> 6, jc = e & 63;
        float ig = shg[(0 * 16 + mm) * 64 + jc];
        float fg = shg[(1 * 16 + mm) * 64 + jc];
        float gg = shg[(2 * 16 + mm) * 64 + jc];
        float og = shg[(3 * 16 + mm) * 64 + jc];
        int b = btile * 16 + mm, jj = jb * 64 + jc;
        float co = cx[b * L_ + jj];
        float cn = sigmoidf_(fg) * co + sigmoidf_(ig) * tanhf(gg);
        cx[b * L_ + jj] = cn;
        hx[b * L_ + jj] = sigmoidf_(og) * tanhf(cn);
      }
      __syncthreads();
    }
    grid_sync(bar);
  }
  if (wg == 0 && tid < 128) {  // pred for t = 99
    int b = tid >> 1, c = tid & 1;
    float s = outB[c];
    for (int l = 0; l < L_; ++l) s += hx[b * L_ + l] * outW[l * 2 + c];
    out_stack[99 * (B_ * 2) + b * 2 + c] = s;
  }
}

// ---------- small prep / epilogue kernels ----------

__global__ void prep_bn(const float* g, const float* be, const float* mn,
                        const float* vr, float* scale, float* shift) {
  int c = blockIdx.x * 256 + threadIdx.x;
  if (c < C_) {
    float s = g[c] * rsqrtf(vr[c] + 1e-5f);
    scale[c] = s;
    shift[c] = be[c] - mn[c] * s;
  }
}
__global__ void prep_wtobj(const float* scale, const float* W, _Float16* Wt) {
  int idx = blockIdx.x * 256 + threadIdx.x;  // 256*1024
  int h = idx >> 10, c = idx & 1023;
  Wt[idx] = (_Float16)(scale[c] * W[c * H_ + h]);
}
__global__ void prep_biasobj(const float* shift, const float* W, const float* ob,
                             float* bo) {
  int h = threadIdx.x;
  float s = ob[h];
  for (int c = 0; c < C_; ++c) s += shift[c] * W[c * H_ + h];
  bo[h] = s;
}
__global__ void prep_t256(const float* S, _Float16* D) {  // D[n*256+h]=S[h*256+n]
  int idx = blockIdx.x * 256 + threadIdx.x;  // 65536
  int n = idx >> 8, h = idx & 255;
  D[idx] = (_Float16)S[h * 256 + n];
}
__global__ void prep_twa(const float* S, _Float16* D) {  // D[h*512+l]=S[l*256+h]
  int idx = blockIdx.x * 256 + threadIdx.x;  // 131072
  int h = idx >> 9, l = idx & 511;
  D[idx] = (_Float16)S[l * H_ + h];
}
__global__ void prep_half(const float* S, _Float16* D, int n) {
  int idx = blockIdx.x * 256 + threadIdx.x;
  if (idx < n) D[idx] = (_Float16)S[idx];
}
__global__ void prep_gsum(const float* a, const float* b, float* d) {
  int i = blockIdx.x * 256 + threadIdx.x;
  if (i < 4 * L_) d[i] = a[i] + b[i];
}
__global__ void frame_avg_k(const float* __restrict__ fr, float* __restrict__ avg) {
  size_t idx = (size_t)blockIdx.x * 256 + threadIdx.x;  // B*T*F = 1638400
  const float* p = fr + idx * 49;
  float s = 0.f;
  for (int i = 0; i < 49; ++i) s += p[i];
  avg[idx] = s * (1.f / 49.f);
}
__global__ void softpred_k(float* out) {
  int idx = blockIdx.x * 256 + threadIdx.x;  // 6400
  if (idx >= B_ * T_) return;
  int b = idx / T_, t = idx % T_;
  const float* ps = out + 140800 + t * (B_ * 2) + b * 2;
  float p0 = ps[0], p1 = ps[1];
  float mx = fmaxf(p0, p1);
  float e0 = __expf(p0 - mx), e1 = __expf(p1 - mx);
  float inv = 1.f / (e0 + e1);
  out[(b * T_ + t) * 2 + 0] = e0 * inv;
  out[(b * T_ + t) * 2 + 1] = e1 * inv;
}

// ---------- host ----------

extern "C" void kernel_launch(void* const* d_in, const int* in_sizes, int n_in,
                              void* d_out, int out_size, void* d_ws, size_t ws_size,
                              hipStream_t stream) {
  const float* feat    = (const float*)d_in[0];
  const float* frame   = (const float*)d_in[1];
  const float* frame_W = (const float*)d_in[2];
  const float* frame_b = (const float*)d_in[3];
  const float* obj_W   = (const float*)d_in[4];
  const float* obj_b   = (const float*)d_in[5];
  const float* gamma   = (const float*)d_in[6];
  const float* beta    = (const float*)d_in[7];
  const float* mean    = (const float*)d_in[8];
  const float* var     = (const float*)d_in[9];
  const float* W_ih    = (const float*)d_in[10];
  const float* W_hh    = (const float*)d_in[11];
  const float* b_ih    = (const float*)d_in[12];
  const float* b_hh    = (const float*)d_in[13];
  const float* out_W   = (const float*)d_in[14];
  const float* out_b   = (const float*)d_in[15];
  const float* att_w   = (const float*)d_in[16];
  const float* att_wa  = (const float*)d_in[17];
  const float* att_ua  = (const float*)d_in[18];
  const float* att_ba  = (const float*)d_in[19];
  float* out = (float*)d_out;

  char* base = (char*)d_ws;
  size_t off = 0;
  auto alloc = [&](size_t bytes) -> void* {
    void* p = base + off;
    off += (bytes + 255) & ~(size_t)255;
    return p;
  };
  _Float16* WtObj = (_Float16*)alloc((size_t)H_ * C_ * 2);       // 256x1024
  _Float16* WtUa  = (_Float16*)alloc((size_t)H_ * H_ * 2);       // 256x256
  _Float16* WtFr  = (_Float16*)alloc((size_t)H_ * F_ * 2);       // 256x256
  _Float16* WihH  = (_Float16*)alloc((size_t)4 * L_ * 2 * H_ * 2);
  _Float16* WhhH  = (_Float16*)alloc((size_t)4 * L_ * L_ * 2);
  _Float16* WwaH  = (_Float16*)alloc((size_t)H_ * L_ * 2);
  float* scale   = (float*)alloc(C_ * 4);
  float* shift   = (float*)alloc(C_ * 4);
  float* biasObj = (float*)alloc(H_ * 4);
  float* gsum    = (float*)alloc(4 * L_ * 4);
  float* favg    = (float*)alloc((size_t)B_ * T_ * F_ * 4);
  float* img     = (float*)alloc((size_t)T_ * B_ * H_ * 4);
  float* objm    = (float*)alloc((size_t)T_ * N_ * B_ * H_ * 4);
  float* ipb     = (float*)alloc((size_t)T_ * N_ * B_ * H_ * 4);
  float* maskb   = (float*)alloc((size_t)T_ * N_ * B_ * 4);
  float* hx      = (float*)alloc((size_t)B_ * L_ * 4);
  float* cx      = (float*)alloc((size_t)B_ * L_ * 4);
  float* hterm   = (float*)alloc((size_t)B_ * 4 * L_ * 4);
  float* pa      = (float*)alloc((size_t)B_ * H_ * 4);
  float* fusion  = (float*)alloc((size_t)B_ * 2 * H_ * 4);
  int*   bar     = (int*)alloc(256);

  hipMemsetAsync(bar, 0, 256, stream);
  hipMemsetAsync(hx, 0, (size_t)B_ * L_ * 4, stream);
  hipMemsetAsync(cx, 0, (size_t)B_ * L_ * 4, stream);

  prep_bn<<<4, 256, 0, stream>>>(gamma, beta, mean, var, scale, shift);
  prep_wtobj<<<(H_ * C_) / 256, 256, 0, stream>>>(scale, obj_W, WtObj);
  prep_biasobj<<<1, 256, 0, stream>>>(shift, obj_W, obj_b, biasObj);
  prep_t256<<<(H_ * H_) / 256, 256, 0, stream>>>(att_ua, WtUa);
  prep_t256<<<(H_ * F_) / 256, 256, 0, stream>>>(frame_W, WtFr);
  prep_twa<<<(H_ * L_) / 256, 256, 0, stream>>>(att_wa, WwaH);
  prep_half<<<(4 * L_ * 2 * H_) / 256, 256, 0, stream>>>(W_ih, WihH, 4 * L_ * 2 * H_);
  prep_half<<<(4 * L_ * L_) / 256, 256, 0, stream>>>(W_hh, WhhH, 4 * L_ * L_);
  prep_gsum<<<8, 256, 0, stream>>>(b_ih, b_hh, gsum);

  frame_avg_k<<<(B_ * T_ * F_) / 256, 256, 0, stream>>>(frame, favg);

  // obj: M=128000, K=1024 ; ip: M=128000, K=256 ; img: M=6400, K=256
  wmma_gemm<0><<<(B_ * T_ * N_ / 16) / 8, 256, 0, stream>>>(feat, WtObj, biasObj,
                                                            objm, maskb, C_, C_);
  wmma_gemm<1><<<(B_ * T_ * N_ / 16) / 8, 256, 0, stream>>>(objm, WtUa, att_ba,
                                                            ipb, nullptr, H_, H_);
  wmma_gemm<2><<<(B_ * T_ / 16) / 8, 256, 0, stream>>>(favg, WtFr, frame_b,
                                                       img, nullptr, F_, F_);

  recurrent_k<<<NWG, 256, 0, stream>>>(ipb, objm, img, maskb, WihH, WhhH, WwaH,
                                       gsum, att_w, out_W, out_b, hx, cx, hterm,
                                       pa, fusion, out + 12800, out + 140800, bar);

  softpred_k<<<(B_ * T_ + 255) / 256, 256, 0, stream>>>(out);
}
